// DCNAlignment_33500744909101
// MI455X (gfx1250) — matrix operand order repaired and checked by reading, not compiled, not run on hardware
//
#include <hip/hip_runtime.h>
#include <hip/hip_bf16.h>
#include <math.h>

// ---------------- problem constants ----------------
#define BD 2
#define TD 7
#define ND 14            // B*T
#define CD 64
#define HD 96
#define WD 96
#define HW (HD*WD)       // 9216
#define MTOT (ND*HW)     // 129024 pixels
#define C2 128           // concat channels
#define CO2 216          // offset conv out channels
#define CO2P 256         // padded
#define GD 8
#define LDSTR 40         // padded LDS row stride (bf16 elems)

typedef __attribute__((ext_vector_type(16))) __bf16 v16bf;
typedef __attribute__((ext_vector_type(8)))  float  v8f;

// ---------------- CDNA5 async global->LDS staging ----------------
#if defined(__has_builtin)
#if __has_builtin(__builtin_amdgcn_global_load_async_to_lds_b128)
#define HAVE_ASYNC 1
#endif
#endif
#ifndef HAVE_ASYNC
#define HAVE_ASYNC 0
#endif

// builtin signature (from hipcc diagnostic): params are pointers to
// 'int __attribute__((vector_size(16)))'
typedef int v4i_vec __attribute__((vector_size(16)));

__device__ __forceinline__ void async_copy16(const void* gsrc, void* ldst) {
#if HAVE_ASYNC
  // global_load_async_to_lds_b128 : LDS[ldst] = MEM[gsrc], tracked by ASYNCcnt
  __builtin_amdgcn_global_load_async_to_lds_b128((v4i_vec*)gsrc, (v4i_vec*)ldst, 0, 0);
#else
  *(uint4*)ldst = *(const uint4*)gsrc;
#endif
}

template <int N>
__device__ __forceinline__ void wait_async() {
#if HAVE_ASYNC
#if __has_builtin(__builtin_amdgcn_s_wait_asynccnt)
  __builtin_amdgcn_s_wait_asynccnt((short)N);
#else
  asm volatile("s_wait_asynccnt %0" ::"i"(N) : "memory");
#endif
#endif
}

__device__ __forceinline__ unsigned short f2bf(float f) {
  unsigned u = __float_as_uint(f);
  unsigned r = u + 0x7FFFu + ((u >> 16) & 1u);   // round-nearest-even
  return (unsigned short)(r >> 16);
}
__device__ __forceinline__ float gelu_exact(float v) {
  return 0.5f * v * (1.0f + erff(v * 0.70710678118654752f));
}
__device__ __forceinline__ float sigmoidf_(float v) {
  return 1.0f / (1.0f + __expf(-v));
}

// ---------------- packing kernels ----------------
__global__ __launch_bounds__(256) void zero_page(unsigned short* __restrict__ z) {
  z[blockIdx.x * blockDim.x + threadIdx.x] = 0;
}

// xcat NHWC bf16: [pix][c2], c2 = [ref 64 | cur 64]
__global__ __launch_bounds__(256) void pack_xcat(const float* __restrict__ x,
                                                 unsigned short* __restrict__ xcat) {
  int tid = blockIdx.x * blockDim.x + threadIdx.x;
  const int total = MTOT * CD;
  if (tid >= total) return;
  int c   = tid / MTOT;          // adjacent threads -> adjacent pixels (coalesced reads)
  int pix = tid - c * MTOT;
  int n   = pix / HW;
  int rem = pix - n * HW;
  int b   = n / TD;
  float ref = x[((size_t)(b * TD) * CD + c) * HW + rem];   // frame t=0 of batch b
  float cur = x[((size_t)n * CD + c) * HW + rem];
  size_t o = (size_t)pix * C2;
  xcat[o + c]      = f2bf(ref);
  xcat[o + CD + c] = f2bf(cur);
}

// w_bottleneck (64,128,3,3) -> w1p[kidx][cout 64][cin 128] bf16
__global__ __launch_bounds__(256) void pack_w1(const float* __restrict__ w,
                                               unsigned short* __restrict__ w1p) {
  int tid = blockIdx.x * blockDim.x + threadIdx.x;
  const int total = 9 * CD * C2;
  if (tid >= total) return;
  int kidx = tid / (CD * C2);
  int r    = tid - kidx * CD * C2;
  int cout = r / C2;
  int cin  = r - cout * C2;
  w1p[((size_t)kidx * CD + cout) * C2 + cin] =
      f2bf(w[((size_t)cout * C2 + cin) * 9 + kidx]);
}

// w_offset (216,64,3,3) -> w2p[kidx][cout 256][cin 64] bf16, zero padded
__global__ __launch_bounds__(256) void pack_w2(const float* __restrict__ w,
                                               unsigned short* __restrict__ w2p) {
  int tid = blockIdx.x * blockDim.x + threadIdx.x;
  const int total = 9 * CO2P * CD;
  if (tid >= total) return;
  int kidx = tid / (CO2P * CD);
  int r    = tid - kidx * CO2P * CD;
  int cout = r / CD;
  int cin  = r - cout * CD;
  float v = (cout < CO2) ? w[((size_t)cout * CD + cin) * 9 + kidx] : 0.0f;
  w2p[((size_t)kidx * CO2P + cout) * CD + cin] = f2bf(v);
}

// ---------------- conv1: implicit GEMM, M=64 x N=64 per block, K=1152 ----------------
// double-buffered LDS, async global->LDS staging, bf16 WMMA, fused exact GELU
__global__ __launch_bounds__(256) void conv1_gelu(const unsigned short* __restrict__ xcat,
                                                  const unsigned short* __restrict__ w1p,
                                                  unsigned short* __restrict__ feat,
                                                  const unsigned short* __restrict__ zpage) {
  __shared__ unsigned short As[2][64 * LDSTR];
  __shared__ unsigned short Bs[2][64 * LDSTR];
  const int tid   = threadIdx.x;
  const int lane  = tid & 31;
  const int wv    = tid >> 5;
  const int mbase = blockIdx.x * 64;
  const int m16   = (wv & 3) * 16;     // wave's M subtile
  const int nhalf = (wv >> 2) * 32;    // wave's N half (2x 16-col tiles)

  // per-thread staging coords: row 0..63, 8-elem chunk 0..3
  const int ar = tid >> 2;
  const int ac = (tid & 3) * 8;
  const int m  = mbase + ar;
  const int n  = m / HW;
  const int rm = m - n * HW;
  const int yy = rm / WD;
  const int xx = rm - yy * WD;

  auto issue = [&](int ks, int buf) {
    const int kidx = ks >> 2;
    const int cin0 = (ks & 3) * 32;
    const int ky = kidx / 3 - 1;
    const int kx = kidx % 3 - 1;
    const int sy = yy + ky, sx = xx + kx;
    const bool inb = (sy >= 0) & (sy < HD) & (sx >= 0) & (sx < WD);
    const unsigned short* ap =
        inb ? xcat + ((size_t)(n * HW + sy * WD + sx) * C2 + cin0 + ac)
            : zpage + ac;                                    // zero halo
    const unsigned short* bp =
        w1p + ((size_t)(kidx * CD + ar) * C2 + cin0 + ac);
    async_copy16(ap, &As[buf][ar * LDSTR + ac]);
    async_copy16(bp, &Bs[buf][ar * LDSTR + ac]);
  };

  v8f acc0 = {}; v8f acc1 = {};
  issue(0, 0);

  for (int ks = 0; ks < 36; ++ks) {          // K = 9*128, 32 per step
    const int cur = ks & 1;
    if (ks + 1 < 36) { issue(ks + 1, cur ^ 1); wait_async<2>(); }
    else             { wait_async<0>(); }
    __syncthreads();                          // all waves' tiles resident

    const unsigned short* Ab = As[cur];
    const unsigned short* Bb = Bs[cur];
    union { uint4 q[2]; v16bf v; } afr, bfr0, bfr1;
    const int arow  = m16 + (lane & 15);
    const int khalf = (lane >> 4) * 8;       // A layout: K {kh..kh+7, kh+16..kh+23}
    afr.q[0] = *(const uint4*)(Ab + arow * LDSTR + khalf);
    afr.q[1] = *(const uint4*)(Ab + arow * LDSTR + khalf + 16);
    const int kb  = (lane >> 4) * 8;         // B layout: K {kb..kb+7, kb+16..kb+23}
    const int bc0 = nhalf + (lane & 15);
    bfr0.q[0] = *(const uint4*)(Bb + bc0 * LDSTR + kb);
    bfr0.q[1] = *(const uint4*)(Bb + bc0 * LDSTR + kb + 16);
    const int bc1 = bc0 + 16;
    bfr1.q[0] = *(const uint4*)(Bb + bc1 * LDSTR + kb);
    bfr1.q[1] = *(const uint4*)(Bb + bc1 * LDSTR + kb + 16);

    acc0 = __builtin_amdgcn_wmma_f32_16x16x32_bf16(false, afr.v, false, bfr0.v,
                                                   (short)0, acc0, false, false);
    acc1 = __builtin_amdgcn_wmma_f32_16x16x32_bf16(false, afr.v, false, bfr1.v,
                                                   (short)0, acc1, false, false);
    __syncthreads();                          // reads done before buffer reuse
  }

  // D layout: VGPR r -> row r + 8*(lane>=16); col = lane&15
  const int srow = mbase + m16 + ((lane >> 4) * 8);
  const int scol = nhalf + (lane & 15);
#pragma unroll
  for (int r = 0; r < 8; ++r) {
    size_t off = (size_t)(srow + r) * CD;
    feat[off + scol]      = f2bf(gelu_exact(acc0[r]));
    feat[off + scol + 16] = f2bf(gelu_exact(acc1[r]));
  }
}

// ---------------- conv2: implicit GEMM, M=64 x N=64 per block (grid.y covers 256), K=576 ----------------
__global__ __launch_bounds__(256) void conv2_offsets(const unsigned short* __restrict__ feat,
                                                     const unsigned short* __restrict__ w2p,
                                                     float* __restrict__ obuf,
                                                     const unsigned short* __restrict__ zpage) {
  __shared__ unsigned short As[2][64 * LDSTR];
  __shared__ unsigned short Bs[2][64 * LDSTR];
  const int tid   = threadIdx.x;
  const int lane  = tid & 31;
  const int wv    = tid >> 5;
  const int mbase = blockIdx.x * 64;
  const int nb    = blockIdx.y * 64;
  const int m16   = (wv & 3) * 16;
  const int nhalf = (wv >> 2) * 32;

  const int ar = tid >> 2;
  const int ac = (tid & 3) * 8;
  const int m  = mbase + ar;
  const int n  = m / HW;
  const int rm = m - n * HW;
  const int yy = rm / WD;
  const int xx = rm - yy * WD;

  auto issue = [&](int ks, int buf) {
    const int kidx = ks >> 1;
    const int cin0 = (ks & 1) * 32;
    const int ky = kidx / 3 - 1;
    const int kx = kidx % 3 - 1;
    const int sy = yy + ky, sx = xx + kx;
    const bool inb = (sy >= 0) & (sy < HD) & (sx >= 0) & (sx < WD);
    const unsigned short* ap =
        inb ? feat + ((size_t)(n * HW + sy * WD + sx) * CD + cin0 + ac)
            : zpage + ac;
    const unsigned short* bp =
        w2p + ((size_t)(kidx * CO2P + nb + ar) * CD + cin0 + ac);
    async_copy16(ap, &As[buf][ar * LDSTR + ac]);
    async_copy16(bp, &Bs[buf][ar * LDSTR + ac]);
  };

  v8f acc0 = {}; v8f acc1 = {};
  issue(0, 0);

  for (int ks = 0; ks < 18; ++ks) {          // K = 9*64, 32 per step
    const int cur = ks & 1;
    if (ks + 1 < 18) { issue(ks + 1, cur ^ 1); wait_async<2>(); }
    else             { wait_async<0>(); }
    __syncthreads();

    const unsigned short* Ab = As[cur];
    const unsigned short* Bb = Bs[cur];
    union { uint4 q[2]; v16bf v; } afr, bfr0, bfr1;
    const int arow  = m16 + (lane & 15);
    const int khalf = (lane >> 4) * 8;
    afr.q[0] = *(const uint4*)(Ab + arow * LDSTR + khalf);
    afr.q[1] = *(const uint4*)(Ab + arow * LDSTR + khalf + 16);
    const int kb  = (lane >> 4) * 8;
    const int bc0 = nhalf + (lane & 15);
    bfr0.q[0] = *(const uint4*)(Bb + bc0 * LDSTR + kb);
    bfr0.q[1] = *(const uint4*)(Bb + bc0 * LDSTR + kb + 16);
    const int bc1 = bc0 + 16;
    bfr1.q[0] = *(const uint4*)(Bb + bc1 * LDSTR + kb);
    bfr1.q[1] = *(const uint4*)(Bb + bc1 * LDSTR + kb + 16);

    acc0 = __builtin_amdgcn_wmma_f32_16x16x32_bf16(false, afr.v, false, bfr0.v,
                                                   (short)0, acc0, false, false);
    acc1 = __builtin_amdgcn_wmma_f32_16x16x32_bf16(false, afr.v, false, bfr1.v,
                                                   (short)0, acc1, false, false);
    __syncthreads();
  }

  const int srow = mbase + m16 + ((lane >> 4) * 8);
  const int col0 = nb + nhalf + (lane & 15);
  const int col1 = col0 + 16;
#pragma unroll
  for (int r = 0; r < 8; ++r) {
    size_t off = (size_t)(srow + r) * CO2;
    if (col0 < CO2) {
      float v = acc0[r];
      obuf[off + col0] = (col0 >= 144) ? sigmoidf_(v) : v;
    }
    if (col1 < CO2) {
      float v = acc1[r];
      obuf[off + col1] = (col1 >= 144) ? sigmoidf_(v) : v;
    }
  }
}

// ---------------- deformable conv (scalar, gather dominated) ----------------
__global__ __launch_bounds__(256) void deform_kernel(const float* __restrict__ x,
                                                     const float* __restrict__ obuf,
                                                     const float* __restrict__ wd,
                                                     const float* __restrict__ bd,
                                                     float* __restrict__ out) {
  int tid = blockIdx.x * blockDim.x + threadIdx.x;
  const int total = MTOT * GD;
  if (tid >= total) return;
  int g   = tid / MTOT;          // adjacent threads -> adjacent pixels
  int pix = tid - g * MTOT;
  int n   = pix / HW;
  int rem = pix - n * HW;
  int y   = rem / WD;
  int xw  = rem - y * WD;
  int t   = n % TD;

  if (t == 0) {                  // frame 0: passthrough
#pragma unroll
    for (int oo = 0; oo < 8; ++oo) {
      size_t idx = ((size_t)n * CD + g * 8 + oo) * HW + rem;
      out[idx] = x[idx];
    }
    return;
  }

  const float* op = obuf + (size_t)pix * CO2;
  float acc[8] = {0.f, 0.f, 0.f, 0.f, 0.f, 0.f, 0.f, 0.f};
  const float* xg = x + ((size_t)n * CD + g * 8) * HW;

#pragma unroll
  for (int k = 0; k < 9; ++k) {
    float dy = op[g * 18 + 2 * k];
    float dx = op[g * 18 + 2 * k + 1];
    float mk = op[144 + g * 9 + k];
    float py = dy + (float)y  + (float)((k / 3) * 2 - 2);   // DIL=2, PAD=2
    float px = dx + (float)xw + (float)((k % 3) * 2 - 2);
    float fy = floorf(py), fx = floorf(px);
    float wy = py - fy, wx = px - fx;
    int iy0 = (int)fy, ix0 = (int)fx;

    float s[8] = {0.f, 0.f, 0.f, 0.f, 0.f, 0.f, 0.f, 0.f};
#pragma unroll
    for (int iy = 0; iy < 2; ++iy) {
#pragma unroll
      for (int ix = 0; ix < 2; ++ix) {
        int yy2 = iy0 + iy, xx2 = ix0 + ix;
        if (yy2 < 0 || yy2 >= HD || xx2 < 0 || xx2 >= WD) continue;
        float wg = (iy ? wy : 1.f - wy) * (ix ? wx : 1.f - wx);
        const float* xp = xg + yy2 * WD + xx2;
#pragma unroll
        for (int c = 0; c < 8; ++c) s[c] = fmaf(wg, xp[(size_t)c * HW], s[c]);
      }
    }
#pragma unroll
    for (int c = 0; c < 8; ++c) s[c] *= mk;
#pragma unroll
    for (int oo = 0; oo < 8; ++oo) {
      const float* wp = wd + ((size_t)(g * 8 + oo) * 8) * 9 + k;   // (64,8,3,3)
#pragma unroll
      for (int c = 0; c < 8; ++c) acc[oo] = fmaf(s[c], wp[c * 9], acc[oo]);
    }
  }
#pragma unroll
  for (int oo = 0; oo < 8; ++oo) {
    out[((size_t)n * CD + g * 8 + oo) * HW + rem] = acc[oo] + bd[g * 8 + oo];
  }
}

// ---------------- host orchestration ----------------
extern "C" void kernel_launch(void* const* d_in, const int* in_sizes, int n_in,
                              void* d_out, int out_size, void* d_ws, size_t ws_size,
                              hipStream_t stream) {
  const float* x   = (const float*)d_in[0];
  const float* wb  = (const float*)d_in[1];
  const float* wo  = (const float*)d_in[2];
  const float* wdp = (const float*)d_in[3];
  const float* bdp = (const float*)d_in[4];
  float* out = (float*)d_out;

  // workspace layout (bytes)
  char* ws = (char*)d_ws;
  constexpr size_t OFF_XCAT = 0;                                  // 14*9216*128 bf16
  constexpr size_t SZ_XCAT  = (size_t)MTOT * C2 * 2;
  constexpr size_t OFF_W1   = OFF_XCAT + SZ_XCAT;                 // 9*64*128 bf16
  constexpr size_t SZ_W1    = (size_t)9 * CD * C2 * 2;
  constexpr size_t OFF_FEAT = OFF_W1 + SZ_W1;                     // 14*9216*64 bf16
  constexpr size_t SZ_FEAT  = (size_t)MTOT * CD * 2;
  constexpr size_t OFF_W2   = OFF_FEAT + SZ_FEAT;                 // 9*256*64 bf16
  constexpr size_t SZ_W2    = (size_t)9 * CO2P * CD * 2;
  constexpr size_t OFF_O    = OFF_W2 + SZ_W2;                     // 14*9216*216 fp32
  constexpr size_t SZ_O     = (size_t)MTOT * CO2 * 4;
  constexpr size_t OFF_Z    = OFF_O + SZ_O;                       // 4KB zero page
  unsigned short* xcat  = (unsigned short*)(ws + OFF_XCAT);
  unsigned short* w1p   = (unsigned short*)(ws + OFF_W1);
  unsigned short* feat  = (unsigned short*)(ws + OFF_FEAT);
  unsigned short* w2p   = (unsigned short*)(ws + OFF_W2);
  float*          obuf  = (float*)(ws + OFF_O);
  unsigned short* zpage = (unsigned short*)(ws + OFF_Z);

  zero_page<<<8, 256, 0, stream>>>(zpage);                        // 4KB zeros
  {
    int total = MTOT * CD;
    pack_xcat<<<(total + 255) / 256, 256, 0, stream>>>(x, xcat);
  }
  {
    int total = 9 * CD * C2;
    pack_w1<<<(total + 255) / 256, 256, 0, stream>>>(wb, w1p);
  }
  {
    int total = 9 * CO2P * CD;
    pack_w2<<<(total + 255) / 256, 256, 0, stream>>>(wo, w2p);
  }
  conv1_gelu<<<MTOT / 64, 256, 0, stream>>>(xcat, w1p, feat, zpage);
  {
    dim3 grid(MTOT / 64, CO2P / 64);
    conv2_offsets<<<grid, 256, 0, stream>>>(feat, w2p, obuf, zpage);
  }
  {
    int total = MTOT * GD;
    deform_kernel<<<(total + 255) / 256, 256, 0, stream>>>(x, obuf, wdp, bdp, out);
  }
}
